// init_59047210385494
// MI455X (gfx1250) — matrix operand order repaired and checked by reading, not compiled, not run on hardware
//
#include <hip/hip_runtime.h>

typedef __attribute__((ext_vector_type(16))) _Float16 v16h;
typedef __attribute__((ext_vector_type(8)))  float    v8f;

union Frag {
    v16h h;
    int4 q[2];
};

#define HD   512   // hidden dim
#define RB   6     // rbf dim
#define MT   128   // edges per block
#define NTC  256   // output columns per block (grid.y = 2)
#define KT   32    // K per WMMA step
#define NKS  (HD / KT)
#define BPAD 264   // padded B row (halves): 528B = 33*16

// ---------------------------------------------------------------------------
__global__ void precompute_tables(const float* __restrict__ emb,
                                  const float* __restrict__ W_lin,
                                  const float* __restrict__ b_lin,
                                  float* __restrict__ T1,
                                  float* __restrict__ T2) {
    int v = blockIdx.x;  // 0..94
    for (int c = threadIdx.x; c < HD; c += blockDim.x) {
        float a1 = b_lin[c];
        float a2 = 0.f;
        for (int k = 0; k < HD; ++k) {
            float ev = emb[v * HD + k];
            a1 += ev * W_lin[(size_t)k * HD + c];
            a2 += ev * W_lin[(size_t)(HD + k) * HD + c];
        }
        T1[v * HD + c] = a1;
        T2[v * HD + c] = a2;
    }
}

__global__ void pack_b(const float* __restrict__ W_lin, _Float16* __restrict__ Bp) {
    int idx = blockIdx.x * 256 + threadIdx.x;  // < 512*512
    Bp[idx] = (_Float16)W_lin[(size_t)(2 * HD) * HD + idx];
}

// ---------------------------------------------------------------------------
__device__ __forceinline__ void async_b128(unsigned ldsoff, unsigned goff,
                                           const _Float16* base) {
    asm volatile("global_load_async_to_lds_b128 %0, %1, %2"
                 :: "v"(ldsoff), "v"(goff), "s"(base)
                 : "memory");
}

// ---------------------------------------------------------------------------
// Block = 256 threads = 8 waves arranged 2(M) x 4(N); wave = 64x64 patch =
// 4x4 WMMA tiles. s_A and s_B double-buffered; A-compute for chunk ks+1
// co-executes with the WMMAs of chunk ks. launch_bounds(256,2): target 2
// blocks/WGP so barriers/async waits of one block hide under the other.
// ---------------------------------------------------------------------------
__global__ __launch_bounds__(256, 2)
void fused_edge_kernel(const float* __restrict__ rbf,
                       const int*   __restrict__ xn,
                       const int*   __restrict__ ei,
                       const int*   __restrict__ ej,
                       const float* __restrict__ W_rbf0,
                       const float* __restrict__ b_rbf0,
                       const float* __restrict__ W_rbf1,
                       const float* __restrict__ T1,
                       const float* __restrict__ T2,
                       const _Float16* __restrict__ Bp,
                       float* __restrict__ e1o,
                       float* __restrict__ e2o,
                       int E) {
    __shared__ __align__(16) float    s_rbf[MT][8];       // 6 used
    __shared__            int      s_r1[MT];
    __shared__            int      s_r2[MT];
    __shared__ __align__(16) _Float16 s_A[2][MT][40];     // 32 used, 80B rows
    __shared__ __align__(16) _Float16 s_B[2][KT][BPAD];

    const int tid  = threadIdx.x;
    const int e0   = blockIdx.x * MT;
    const int nb   = blockIdx.y * NTC;
    const int lane = tid & 31;
    const int wv   = tid >> 5;
    const int mg   = wv & 1;   // M-group: rows [mg*64, +64)
    const int ng   = wv >> 1;  // N-group: cols [ng*64, +64)

    // ---- stage per-edge metadata ----
    if (tid < MT) {
        int  e     = e0 + tid;
        bool valid = (e < E);
        int  vi    = valid ? ei[e] : 0;
        int  vj    = valid ? ej[e] : 0;
        s_r1[tid]  = valid ? xn[vi] : 0;
        s_r2[tid]  = valid ? xn[vj] : 0;
#pragma unroll
        for (int r = 0; r < RB; ++r)
            s_rbf[tid][r] = valid ? rbf[(size_t)e * RB + r] : 0.f;
    }

    const _Float16* bbase = Bp + nb;  // uniform

    // chunk = 32 rows x 256 halves = 1024 x 16B segments, 4 per thread
#define ISSUE_CHUNK(ks_, buf_)                                                  \
    {                                                                           \
        _Pragma("unroll")                                                       \
        for (int q = 0; q < 4; ++q) {                                           \
            int s_  = tid + 256 * q;                                            \
            int row = s_ >> 5;                                                  \
            int sc  = s_ & 31;                                                  \
            unsigned ldsoff = (unsigned)(size_t)&s_B[buf_][row][sc * 8];        \
            unsigned goff = (unsigned)((((ks_) * KT + row) * HD + sc * 8) * 2); \
            async_b128(ldsoff, goff, bbase);                                    \
        }                                                                       \
    }

    ISSUE_CHUNK(0, 0)
    __syncthreads();  // metadata visible

    // this thread's A assignment: one edge, 16 contiguous k per step
    const int eloc_a  = tid & (MT - 1);
    const int ka_base = (tid >> 7) * 16;  // 0 or 16 (wave-uniform)
    float rb[RB];
#pragma unroll
    for (int r = 0; r < RB; ++r) rb[r] = s_rbf[eloc_a][r];

    // A-chunk: swish(rbf @ W_rbf0 + b); k wave-uniform -> scalar loads.
    // W_rbf0 streamed row-by-row to keep register liveness short.
    auto computeA = [&](int ksA, int buf) {
#pragma unroll
        for (int g8 = 0; g8 < 2; ++g8) {
            int kk0 = __builtin_amdgcn_readfirstlane(ksA * KT + ka_base + g8 * 8);
            float z[8];
#pragma unroll
            for (int u = 0; u < 8; ++u) z[u] = b_rbf0[kk0 + u];
#pragma unroll
            for (int r = 0; r < RB; ++r) {
                float wrow[8];
#pragma unroll
                for (int u = 0; u < 8; ++u) wrow[u] = W_rbf0[r * HD + kk0 + u];
#pragma unroll
                for (int u = 0; u < 8; ++u) z[u] += rb[r] * wrow[u];
            }
            _Float16 vals[8];
#pragma unroll
            for (int u = 0; u < 8; ++u)
                vals[u] = (_Float16)(z[u] * __builtin_amdgcn_rcpf(1.f + __expf(-z[u])));
            *(int4*)&s_A[buf][eloc_a][ka_base + g8 * 8] = *(const int4*)vals;
        }
    };

    computeA(0, 0);  // prologue chunk (visible after step-0 barrier)

    v8f acc[16] = {};
    const int m  = lane & 15;
    const int kb = (lane < 16) ? 0 : 8;

    for (int ks = 0; ks < NKS; ++ks) {
        if (ks + 1 < NKS) ISSUE_CHUNK(ks + 1, (ks + 1) & 1)

        if (ks + 1 < NKS) asm volatile("s_wait_asynccnt 0x4" ::: "memory");
        else              asm volatile("s_wait_asynccnt 0x0" ::: "memory");
        __syncthreads();  // s_A[ks&1] + s_B[ks&1] visible

        const _Float16(*sa)[40]   = s_A[ks & 1];
        const _Float16(*sb)[BPAD] = s_B[ks & 1];
        Frag a[4];
#pragma unroll
        for (int mt = 0; mt < 4; ++mt) {
            int row = mg * 64 + mt * 16 + m;
            a[mt].q[0] = *(const int4*)&sa[row][kb];
            a[mt].q[1] = *(const int4*)&sa[row][kb + 16];
        }
#pragma unroll
        for (int nt = 0; nt < 4; ++nt) {
            Frag b;
            int cb = (ng * 4 + nt) * 16;
#pragma unroll
            for (int u = 0; u < 16; ++u)
                b.h[u] = sb[(u < 8) ? (kb + u) : (16 + kb + u - 8)][cb + m];
#pragma unroll
            for (int mt = 0; mt < 4; ++mt)
                acc[mt * 4 + nt] = __builtin_amdgcn_wmma_f32_16x16x32_f16(
                    false, a[mt].h, false, b.h, (short)0, acc[mt * 4 + nt],
                    false, false);
        }

        // co-executes with the WMMAs above (separate pipes, no barrier between)
        if (ks + 1 < NKS) computeA(ks + 1, (ks + 1) & 1);

        __syncthreads();  // buffers consumed / next A staged
    }

    // ---- K=32 fragments for g = rbf @ W_rbf1 (rows/cols >= 6 are zero) ----
    Frag a2[4];
#pragma unroll
    for (int mt = 0; mt < 4; ++mt) {
        int row = mg * 64 + mt * 16 + m;
        _Float16 v[16];
#pragma unroll
        for (int u = 0; u < 16; ++u) v[u] = (_Float16)0.f;
        if (lane < 16) {  // lanes<16 hold K=0..7 in q[0]
#pragma unroll
            for (int r = 0; r < RB; ++r) v[r] = (_Float16)s_rbf[row][r];
        }
        a2[mt].q[0] = ((const int4*)v)[0];
        a2[mt].q[1] = ((const int4*)v)[1];
    }
    Frag b2[4];
#pragma unroll
    for (int nt = 0; nt < 4; ++nt) {
        int cb = nb + (ng * 4 + nt) * 16;
        float tmp[16];
#pragma unroll
        for (int u = 0; u < 16; ++u) tmp[u] = 0.f;
        if (lane < 16) {  // B: lane = N col, element u = K; only K<6 nonzero
#pragma unroll
            for (int u = 0; u < RB; ++u) tmp[u] = W_rbf1[u * HD + cb + lane];
        }
        _Float16 v[16];
#pragma unroll
        for (int u = 0; u < 16; ++u) v[u] = (_Float16)tmp[u];
        b2[nt].q[0] = ((const int4*)v)[0];
        b2[nt].q[1] = ((const int4*)v)[1];
    }

    // ---- epilogue ----
    const int  hf    = lane >> 4;
    const int  nlo   = lane & 15;
    const v8f  zero8 = {};
    const bool full  = (e0 + MT <= E);  // block-uniform

    if (full) {
#pragma unroll
        for (int nt = 0; nt < 4; ++nt) {
            int c = nb + (ng * 4 + nt) * 16 + nlo;
#pragma unroll
            for (int mt = 0; mt < 4; ++mt) {
                v8f g = __builtin_amdgcn_wmma_f32_16x16x32_f16(
                    false, a2[mt].h, false, b2[nt].h, (short)0, zero8, false, false);
                v8f av = acc[mt * 4 + nt];
#pragma unroll
                for (int p = 0; p < 8; ++p) {
                    int   eloc = mg * 64 + mt * 16 + hf * 8 + p;
                    int   e    = e0 + eloc;
                    float pre  = av[p] + T1[s_r1[eloc] * HD + c]
                                       + T2[s_r2[eloc] * HD + c];
                    float e1v  = pre * __builtin_amdgcn_rcpf(1.f + __expf(-pre));
                    __builtin_nontemporal_store(e1v, &e1o[(size_t)e * HD + c]);
                    __builtin_nontemporal_store(g[p] * e1v, &e2o[(size_t)e * HD + c]);
                }
            }
        }
    } else {
#pragma unroll
        for (int nt = 0; nt < 4; ++nt) {
            int c = nb + (ng * 4 + nt) * 16 + nlo;
#pragma unroll
            for (int mt = 0; mt < 4; ++mt) {
                v8f g = __builtin_amdgcn_wmma_f32_16x16x32_f16(
                    false, a2[mt].h, false, b2[nt].h, (short)0, zero8, false, false);
                v8f av = acc[mt * 4 + nt];
#pragma unroll
                for (int p = 0; p < 8; ++p) {
                    int eloc = mg * 64 + mt * 16 + hf * 8 + p;
                    int e    = e0 + eloc;
                    if (e < E) {
                        float pre = av[p] + T1[s_r1[eloc] * HD + c]
                                          + T2[s_r2[eloc] * HD + c];
                        float e1v = pre * __builtin_amdgcn_rcpf(1.f + __expf(-pre));
                        __builtin_nontemporal_store(e1v, &e1o[(size_t)e * HD + c]);
                        __builtin_nontemporal_store(g[p] * e1v, &e2o[(size_t)e * HD + c]);
                    }
                }
            }
        }
    }
#undef ISSUE_CHUNK
}

// ---------------------------------------------------------------------------
extern "C" void kernel_launch(void* const* d_in, const int* in_sizes, int n_in,
                              void* d_out, int out_size, void* d_ws, size_t ws_size,
                              hipStream_t stream) {
    const int*   x      = (const int*)d_in[0];
    const float* rbf    = (const float*)d_in[1];
    const int*   ei     = (const int*)d_in[2];
    const int*   ej     = (const int*)d_in[3];
    const float* emb    = (const float*)d_in[4];
    const float* W_rbf0 = (const float*)d_in[5];
    const float* b_rbf0 = (const float*)d_in[6];
    const float* W_lin  = (const float*)d_in[7];
    const float* b_lin  = (const float*)d_in[8];
    const float* W_rbf1 = (const float*)d_in[9];

    const int E = in_sizes[2];

    float*    T1 = (float*)d_ws;
    float*    T2 = T1 + 95 * HD;
    _Float16* Bp = (_Float16*)(T2 + 95 * HD);

    float* e1o = (float*)d_out;
    float* e2o = e1o + (size_t)E * HD;

    precompute_tables<<<95, 256, 0, stream>>>(emb, W_lin, b_lin, T1, T2);
    pack_b<<<(HD * HD) / 256, 256, 0, stream>>>(W_lin, Bp);

    dim3 grid((E + MT - 1) / MT, HD / NTC);
    fused_edge_kernel<<<grid, 256, 0, stream>>>(rbf, x, ei, ej, W_rbf0, b_rbf0,
                                                W_rbf1, T1, T2, Bp, e1o, e2o, E);
}
